// MultiHeadAttention_88785563943252
// MI455X (gfx1250) — matrix operand order repaired and checked by reading, not compiled, not run on hardware
//
#include <hip/hip_runtime.h>

typedef __attribute__((ext_vector_type(16))) _Float16 v16h;
typedef __attribute__((ext_vector_type(8)))  float    v8f;
typedef __attribute__((ext_vector_type(4)))  unsigned int u32x4;
typedef __attribute__((ext_vector_type(8)))  int      i32x8;
typedef __attribute__((ext_vector_type(4)))  int      i32x4;

#define EMB   1024
#define NHEAD 16
#define DHEAD 64
#define BATCH 4
#define SEQ   2048

// ---------------------------------------------------------------------------
// Fragment loaders for CDNA5 WMMA 16-bit operands; this file uses contiguous k (same in A and B)
// A (16x32, MxK): lane L holds row M=L&15.  lanes 0-15: K=0..15;
// lanes 16-31: K=16..31  => two contiguous 8-half (16B) loads.
// B (32x16, KxN): lane L holds column N=L&15; lanes 0-15 hold K=0..15, lanes
// 16-31 hold K=16..31 => one contiguous 16-half run at k = hi*16.
// C/D (16x16 f32): VGPR r, lane L -> M = r + 8*(L>>4), N = L&15.
// ---------------------------------------------------------------------------
static __device__ __forceinline__ v16h load_frag_a(const _Float16* p) {
    v16h r;
    ((float4*)&r)[0] = *(const float4*)(p);
    ((float4*)&r)[1] = *(const float4*)(p + 8);
    return r;
}
static __device__ __forceinline__ v16h load_frag_b(const _Float16* p) {
    v16h r;
    ((float4*)&r)[0] = *(const float4*)(p);
    ((float4*)&r)[1] = *(const float4*)(p + 8);
    return r;
}
// B operand gathered from row-major [K][N=64] LDS tile (stride 64 halves).
static __device__ __forceinline__ v16h load_frag_b_stride64(const _Float16* p) {
    v16h r;
#pragma unroll
    for (int j = 0; j < 16; ++j) r[j] = p[j * 64];
    return r;
}
static __device__ __forceinline__ v8f vzero8() {
    v8f z;
    for (int i = 0; i < 8; ++i) z[i] = 0.0f;
    return z;
}
static __device__ __forceinline__ v8f wmma_f16(v16h a, v16h b, v8f c) {
    return __builtin_amdgcn_wmma_f32_16x16x32_f16(false, a, false, b,
                                                  (short)0, c, false, false);
}

// ---------------------------------------------------------------------------
// Tensor Data Mover: DMA a contiguous 8KB (4096 x f16) run from global to LDS.
// D# packed per ISA ch.8.3/8.4: group0 = {count=1, lds_addr, global_addr,
// type=2}; group1 = {data_size=2B, tensor_dim0=4096, tensor_dim1=1,
// tile_dim0=4096, tile_dim1=1, stride0=4096, stride1=4096}; remaining groups
// zero (unused for a 1-row 2D tile).  Tracked with TENSORcnt; in-order
// completion per wave.  clang-23 toolchain: 6-arg builtin form.
// ---------------------------------------------------------------------------
static __device__ __forceinline__ void tdm_load_8kb(unsigned lds_byte_off,
                                                    const _Float16* gptr) {
    unsigned long long ga = (unsigned long long)gptr;
    u32x4 g0;
    g0[0] = 1u;                                              // count = 1
    g0[1] = lds_byte_off;                                    // LDS dest
    g0[2] = (unsigned)(ga & 0xFFFFFFFFu);                    // global_addr lo
    g0[3] = (unsigned)((ga >> 32) & 0x01FFFFFFu) | 0x80000000u; // hi | type=2
    i32x8 g1;
    g1[0] = 0x00010000;            // workgroup_mask=0, data_size=1 (2 bytes)
    g1[1] = (int)(0x1000u << 16);  // tensor_dim0[15:0]=4096 in bits 63:48
    g1[2] = 0x00010000;            // tensor_dim0 hi=0, tensor_dim1=1
    g1[3] = (int)(0x1000u << 16);  // tensor_dim1 hi=0, tile_dim0=4096
    g1[4] = 0x00000001;            // tile_dim1=1, tile_dim2=0
    g1[5] = 0x00001000;            // tensor_dim0_stride lo = 4096
    g1[6] = (int)(0x1000u << 16);  // stride0 hi=0, tensor_dim1_stride lo=4096
    g1[7] = 0;
    i32x4 z4;
    z4[0] = 0; z4[1] = 0; z4[2] = 0; z4[3] = 0;
    i32x8 z8;
    for (int i = 0; i < 8; ++i) z8[i] = 0;
    __builtin_amdgcn_tensor_load_to_lds(g0, g1, z4, z4, z8, 0);
}

// ---------------------------------------------------------------------------
// f32 -> f16 conversion (vectorized, grid-stride)
// ---------------------------------------------------------------------------
__global__ void cvt_f32_f16_kernel(const float4* __restrict__ in,
                                   _Float16* __restrict__ out, long n4) {
    long i  = (long)blockIdx.x * blockDim.x + threadIdx.x;
    long st = (long)gridDim.x * blockDim.x;
    for (; i < n4; i += st) {
        float4 v = in[i];
        __align__(8) _Float16 h[4] = {(_Float16)v.x, (_Float16)v.y,
                                      (_Float16)v.z, (_Float16)v.w};
        *(unsigned long long*)(out + i * 4) = *(const unsigned long long*)h;
    }
}

// ---------------------------------------------------------------------------
// QKV projection: Out[b,h,s,d] = X[m,:] . W[n,:] + bias[n], f16 out.
// Block = 256 threads = 8 waves (2 M x 4 N); block tile 64 x 128; each wave
// computes 32x32 via 2x2 WMMA fragments, K-steps of 32, operands streamed
// straight from global (whole working set is L2-resident).
// ---------------------------------------------------------------------------
__global__ __launch_bounds__(256) void qkv_proj_kernel(
    const _Float16* __restrict__ X,
    const _Float16* __restrict__ W,
    const float*    __restrict__ bias,
    _Float16* __restrict__ Out) {
    const int lane = threadIdx.x & 31;
    const int wave = threadIdx.x >> 5;
    const int wm   = wave >> 2;          // 0..1
    const int wn   = wave & 3;           // 0..3
    const int hi   = lane >> 4;
    const int ln   = lane & 15;
    const int m0   = blockIdx.x * 64  + wm * 32;
    const int n0   = blockIdx.y * 128 + wn * 32;

    v8f acc[2][2];
    acc[0][0] = vzero8(); acc[0][1] = vzero8();
    acc[1][0] = vzero8(); acc[1][1] = vzero8();

    const _Float16* Ar0 = X + (size_t)(m0 +      ln) * EMB + hi * 16;
    const _Float16* Ar1 = X + (size_t)(m0 + 16 + ln) * EMB + hi * 16;
    const _Float16* Br0 = W + (size_t)(n0 +      ln) * EMB + hi * 16;
    const _Float16* Br1 = W + (size_t)(n0 + 16 + ln) * EMB + hi * 16;

    for (int kk = 0; kk < EMB; kk += 32) {
        v16h a0 = load_frag_a(Ar0 + kk);
        v16h a1 = load_frag_a(Ar1 + kk);
        v16h b0 = load_frag_b(Br0 + kk);
        v16h b1 = load_frag_b(Br1 + kk);
        acc[0][0] = wmma_f16(a0, b0, acc[0][0]);
        acc[0][1] = wmma_f16(a0, b1, acc[0][1]);
        acc[1][0] = wmma_f16(a1, b0, acc[1][0]);
        acc[1][1] = wmma_f16(a1, b1, acc[1][1]);
    }

    for (int fm = 0; fm < 2; ++fm) {
        for (int fn = 0; fn < 2; ++fn) {
            const int n  = n0 + fn * 16 + ln;
            const int h  = n >> 6;
            const int d  = n & 63;
            const float bv = bias[n];
            for (int r = 0; r < 8; ++r) {
                const int m = m0 + fm * 16 + r + 8 * hi;
                const int b = m >> 11;          // / SEQ
                const int s = m & (SEQ - 1);
                Out[(((size_t)b * NHEAD + h) * SEQ + s) * DHEAD + d] =
                    (_Float16)(acc[fm][fn][r] + bv);
            }
        }
    }
}

// ---------------------------------------------------------------------------
// Flash attention, TDM double-buffered K/V staging.
// One block per (b*H, 64-query tile); 8 waves arranged 4(Mx16) x 2(Nx32).
// Wave 0 drives the Tensor Data Mover: prefetch chunk i+1 while all waves
// run WMMA on chunk i; s_wait_tensorcnt(2) leaves only the prefetch pending.
// ---------------------------------------------------------------------------
__global__ __launch_bounds__(256) void attn_kernel(
    const _Float16* __restrict__ Qb,   // [B,H,S,64] f16
    const _Float16* __restrict__ Kb,
    const _Float16* __restrict__ Vb,
    const unsigned char* __restrict__ mask,  // [B,S,S] bool
    float* __restrict__ out) {               // [B,S,H*64] f32
    __shared__ __align__(16) _Float16 Qs[64 * 64];
    __shared__ __align__(16) _Float16 Ks[2][64 * 64];
    __shared__ __align__(16) _Float16 Vs[2][64 * 64];   // raw [key][d]
    __shared__ __align__(16) _Float16 Ps[64 * 64];
    __shared__ __align__(16) float    Ss[64 * 68];      // padded (bank skew)
    __shared__ float mrow[64], lrow[64], arow[64];

    const int tid  = threadIdx.x;
    const int lane = tid & 31;
    const int wave = tid >> 5;
    const int wm   = wave >> 1;   // 0..3
    const int wn   = wave & 1;    // 0..1
    const int hi   = lane >> 4;
    const int ln   = lane & 15;
    const int q0   = blockIdx.x * 64;
    const int bh   = blockIdx.y;
    const int b    = bh / NHEAD;
    const int h    = bh % NHEAD;

    const _Float16* Qg = Qb + ((size_t)bh * SEQ + q0) * DHEAD;
    const _Float16* Kg = Kb + (size_t)bh * SEQ * DHEAD;
    const _Float16* Vg = Vb + (size_t)bh * SEQ * DHEAD;

    const unsigned qs_off = (unsigned)(uintptr_t)(void*)Qs;
    unsigned ks_off[2] = {(unsigned)(uintptr_t)(void*)Ks[0],
                          (unsigned)(uintptr_t)(void*)Ks[1]};
    unsigned vs_off[2] = {(unsigned)(uintptr_t)(void*)Vs[0],
                          (unsigned)(uintptr_t)(void*)Vs[1]};

    // Kick off Q + first K/V chunk through the TDM.
    if (wave == 0) {
        tdm_load_8kb(qs_off,    Qg);
        tdm_load_8kb(ks_off[0], Kg);
        tdm_load_8kb(vs_off[0], Vg);
    }
    if (tid < 64) { mrow[tid] = -3.0e38f; lrow[tid] = 0.0f; }

    v8f o[2];
    o[0] = vzero8(); o[1] = vzero8();

    int buf = 0;
    for (int kc = 0; kc < SEQ; kc += 64) {
        __syncthreads();   // nobody still reading buf^1 (being overwritten)
        if (wave == 0) {
            if (kc + 64 < SEQ) {
                tdm_load_8kb(ks_off[buf ^ 1], Kg + (size_t)(kc + 64) * DHEAD);
                tdm_load_8kb(vs_off[buf ^ 1], Vg + (size_t)(kc + 64) * DHEAD);
                __builtin_amdgcn_s_wait_tensorcnt(2);  // current chunk landed
            } else {
                __builtin_amdgcn_s_wait_tensorcnt(0);
            }
        }
        __syncthreads();   // current K/V (and Q on iter 0) visible to all

        // S = Q . K^T * 1/sqrt(64); mask; stash in Ss
        const _Float16* Kc = Ks[buf];
        for (int fn = 0; fn < 2; ++fn) {
            const int kb = wn * 32 + fn * 16;
            v8f s = vzero8();
            for (int ks = 0; ks < 2; ++ks) {
                v16h a  = load_frag_a(Qs + (wm * 16 + ln) * 64 + ks * 32 + hi * 16);
                v16h bm = load_frag_b(Kc + (kb + ln) * 64 + ks * 32 + hi * 16);
                s = wmma_f16(a, bm, s);
            }
            const int kcol = kb + ln;
            for (int r = 0; r < 8; ++r) {
                const int qrow = wm * 16 + r + 8 * hi;
                float val = s[r] * 0.125f;
                if (mask[((size_t)b * SEQ + (q0 + qrow)) * SEQ + (kc + kcol)])
                    val = -1.0e9f;
                Ss[qrow * 68 + kcol] = val;
            }
        }
        __syncthreads();

        // Online softmax: one thread per query row (skewed column order)
        if (tid < 64) {
            const int row = tid;
            float mx = -3.0e38f;
            for (int c0 = 0; c0 < 64; ++c0) {
                int c = (c0 + row) & 63;
                mx = fmaxf(mx, Ss[row * 68 + c]);
            }
            const float mold = mrow[row];
            const float mnew = fmaxf(mold, mx);
            const float al   = __expf(mold - mnew);
            float l = lrow[row] * al;
            for (int c0 = 0; c0 < 64; ++c0) {
                int c   = (c0 + row) & 63;
                float p = __expf(Ss[row * 68 + c] - mnew);
                Ps[row * 64 + c] = (_Float16)p;
                l += p;
            }
            mrow[row] = mnew; lrow[row] = l; arow[row] = al;
        }
        __syncthreads();

        // Rescale running O, accumulate O += P . V
        float alv[8];
        for (int r = 0; r < 8; ++r) alv[r] = arow[wm * 16 + r + 8 * hi];
        for (int fn = 0; fn < 2; ++fn)
            for (int r = 0; r < 8; ++r) o[fn][r] *= alv[r];
        const _Float16* Vc = Vs[buf];
        for (int fn = 0; fn < 2; ++fn) {
            const int nb = wn * 32 + fn * 16;
            for (int ks = 0; ks < 2; ++ks) {
                v16h a  = load_frag_a(Ps + (wm * 16 + ln) * 64 + ks * 32 + hi * 16);
                v16h bm = load_frag_b_stride64(Vc + (ks * 32 + hi * 16) * 64 + nb + ln);
                o[fn] = wmma_f16(a, bm, o[fn]);
            }
        }
        buf ^= 1;
    }

    // Epilogue: O / l, store [b, q, h*64 + d] f32
    for (int fn = 0; fn < 2; ++fn) {
        const int d = wn * 32 + fn * 16 + ln;
        for (int r = 0; r < 8; ++r) {
            const int qrow = wm * 16 + r + 8 * hi;
            const float inv = 1.0f / lrow[qrow];
            out[((size_t)b * SEQ + (q0 + qrow)) * (NHEAD * DHEAD) + h * DHEAD + d] =
                o[fn][r] * inv;
        }
    }
}

// ---------------------------------------------------------------------------
extern "C" void kernel_launch(void* const* d_in, const int* in_sizes, int n_in,
                              void* d_out, int out_size, void* d_ws, size_t ws_size,
                              hipStream_t stream) {
    (void)in_sizes; (void)n_in; (void)out_size; (void)ws_size;
    const float* x_q  = (const float*)d_in[0];
    const float* x_kv = (const float*)d_in[1];
    const unsigned char* mask = (const unsigned char*)d_in[2];
    const float* w_q = (const float*)d_in[3];
    const float* b_q = (const float*)d_in[4];
    const float* w_k = (const float*)d_in[5];
    const float* b_k = (const float*)d_in[6];
    const float* w_v = (const float*)d_in[7];
    const float* b_v = (const float*)d_in[8];
    float* out = (float*)d_out;

    const size_t NX = (size_t)BATCH * SEQ * EMB;   // 8,388,608
    const size_t NW = (size_t)EMB * EMB;           // 1,048,576
    _Float16* ws  = (_Float16*)d_ws;               // ~90 MB total
    _Float16* Xq  = ws;
    _Float16* Xkv = Xq  + NX;
    _Float16* Wq  = Xkv + NX;
    _Float16* Wk  = Wq  + NW;
    _Float16* Wv  = Wk  + NW;
    _Float16* Qb  = Wv  + NW;
    _Float16* Kb  = Qb  + NX;
    _Float16* Vb  = Kb  + NX;

    cvt_f32_f16_kernel<<<2048, 256, 0, stream>>>((const float4*)x_q,  Xq,  (long)(NX / 4));
    cvt_f32_f16_kernel<<<2048, 256, 0, stream>>>((const float4*)x_kv, Xkv, (long)(NX / 4));
    cvt_f32_f16_kernel<<<512,  256, 0, stream>>>((const float4*)w_q,  Wq,  (long)(NW / 4));
    cvt_f32_f16_kernel<<<512,  256, 0, stream>>>((const float4*)w_k,  Wk,  (long)(NW / 4));
    cvt_f32_f16_kernel<<<512,  256, 0, stream>>>((const float4*)w_v,  Wv,  (long)(NW / 4));

    dim3 gp((BATCH * SEQ) / 64, (NHEAD * DHEAD) / 128);   // (128, 8)
    qkv_proj_kernel<<<gp, 256, 0, stream>>>(Xq,  Wq, b_q, Qb);
    qkv_proj_kernel<<<gp, 256, 0, stream>>>(Xkv, Wk, b_k, Kb);
    qkv_proj_kernel<<<gp, 256, 0, stream>>>(Xkv, Wv, b_v, Vb);

    dim3 ga(SEQ / 64, BATCH * NHEAD);                     // (32, 64)
    attn_kernel<<<ga, 256, 0, stream>>>(Qb, Kb, Vb, mask, out);
}
